// CusMultiHeadAttention_4569845203022
// MI455X (gfx1250) — matrix operand order repaired and checked by reading, not compiled, not run on hardware
//
#include <hip/hip_runtime.h>
#include <hip/hip_bf16.h>

// ---------------------------------------------------------------------------
// CDNA5 (gfx1250) fused multi-head attention, bf16 WMMA, wave32,
// async global->LDS tile fills with ping-pong double buffering.
// ---------------------------------------------------------------------------

typedef __bf16 bf16_t;
typedef __attribute__((ext_vector_type(16))) __bf16 v16bf;
typedef __attribute__((ext_vector_type(8)))  __bf16 v8bf;
typedef __attribute__((ext_vector_type(4)))  __bf16 v4bf;
typedef __attribute__((ext_vector_type(8)))  float  v8f;

#define C_IN   1024
#define NHEAD  16
#define HDIM   64
#define BATCH  4
#define SEQ    1024
#define MTOT   (BATCH*SEQ)   // 4096 rows

__device__ __forceinline__ v8f v8f_zero() {
  v8f z = {0.f,0.f,0.f,0.f,0.f,0.f,0.f,0.f};
  return z;
}

__device__ __forceinline__ v16bf cat8(v8bf lo, v8bf hi) {
  return __builtin_shufflevector(lo, hi, 0,1,2,3,4,5,6,7,8,9,10,11,12,13,14,15);
}

// A fragment (16x32 bf16, M = lane&15): element i -> K = k0 + (i<8?0:16) + half*8 + (i&7)
__device__ __forceinline__ v16bf frag_a(const bf16_t* rowp, int k0, int half) {
  v8bf lo = *(const v8bf*)(rowp + k0 + half*8);
  v8bf hi = *(const v8bf*)(rowp + k0 + 16 + half*8);
  return cat8(lo, hi);
}

// B fragment (32x16 bf16, N = lane&15): element i -> K = k0 + (i<8?0:16) + half*8 + (i&7)
__device__ __forceinline__ v16bf frag_b(const bf16_t* rowp, int k0, int half) {
  v8bf lo = *(const v8bf*)(rowp + k0 + half*8);
  v8bf hi = *(const v8bf*)(rowp + k0 + 16 + half*8);
  return cat8(lo, hi);
}

__device__ __forceinline__ v8f wmma_bf16(v16bf a, v16bf b, v8f c) {
  return __builtin_amdgcn_wmma_f32_16x16x32_bf16(false, a, false, b, (short)0, c,
                                                 false, false);
}

// --- CDNA5 async global->LDS copy (16 bytes / lane), ASYNCcnt tracked -------
// Builtin signature (from hipcc diagnostic): (int4 AS1* src, int4 AS3* dst,
// imm offset, imm cpol).
typedef __attribute__((__vector_size__(4 * sizeof(int)))) int i32x4;

__device__ __forceinline__ void async_cp16(bf16_t* lds_dst, const bf16_t* gsrc) {
#if __has_builtin(__builtin_amdgcn_global_load_async_to_lds_b128)
  __attribute__((address_space(1))) i32x4* src =
      (__attribute__((address_space(1))) i32x4*)
      ((__attribute__((address_space(1))) void*)(void*)(const void*)gsrc);
  __attribute__((address_space(3))) i32x4* dst =
      (__attribute__((address_space(3))) i32x4*)
      ((__attribute__((address_space(3))) void*)(void*)lds_dst);
  __builtin_amdgcn_global_load_async_to_lds_b128(src, dst, 0, 0);
#else
  *(v8bf*)lds_dst = *(const v8bf*)gsrc;   // safe fallback: VGPR round trip
#endif
}

__device__ __forceinline__ void wait_async() {
#if __has_builtin(__builtin_amdgcn_s_wait_asynccnt)
  __builtin_amdgcn_s_wait_asynccnt(0);
#else
  asm volatile("s_wait_asynccnt 0x0" ::: "memory");
#endif
}

// ---------------------------------------------------------------------------
// f32 -> bf16 elementwise convert (x)
// ---------------------------------------------------------------------------
__global__ __launch_bounds__(256)
void k_cvt_x(const float* __restrict__ x, bf16_t* __restrict__ y) {
  int i = (blockIdx.x * 256 + threadIdx.x) * 4;
  float4 v = *(const float4*)(x + i);
  v4bf o = { (__bf16)v.x, (__bf16)v.y, (__bf16)v.z, (__bf16)v.w };
  *(v4bf*)(y + i) = o;
}

// ---------------------------------------------------------------------------
// 1024x1024 f32 -> transposed bf16 (weights -> (N,K) layout)
// ---------------------------------------------------------------------------
__global__ __launch_bounds__(256)
void k_transpose_cvt(const float* __restrict__ W, bf16_t* __restrict__ Wt) {
  __shared__ float tile[32][33];
  const int bx = blockIdx.x & 31, by = blockIdx.x >> 5;
  const int tx = threadIdx.x & 31, ty = threadIdx.x >> 5;
#pragma unroll
  for (int i = 0; i < 32; i += 8)
    tile[ty + i][tx] = W[(size_t)(by*32 + ty + i)*C_IN + bx*32 + tx];
  __syncthreads();
#pragma unroll
  for (int i = 0; i < 32; i += 8)
    Wt[(size_t)(bx*32 + ty + i)*C_IN + by*32 + tx] = (__bf16)tile[tx][ty + i];
}

// ---------------------------------------------------------------------------
// bf16 WMMA GEMM: A(M x 1024 row-major) * Bt(N x 1024 row-major)^T
// mode 0: out bf16 split-head (B,H,S,c), scaled (used for Q/K/V)
// mode 1: out f32 row-major + bias (final projection)
// block = 256 threads (8 waves), tile 128x128, K-step 32,
// double-buffered LDS filled by async global->LDS copies.
// ---------------------------------------------------------------------------
__global__ __launch_bounds__(256)
void k_gemm_bf16(const bf16_t* __restrict__ A, const bf16_t* __restrict__ Bt,
                 void* __restrict__ outp, const float* __restrict__ obias,
                 float scale, int mode) {
  __shared__ bf16_t As[2][128][40];
  __shared__ bf16_t Bs[2][128][40];
  const int rowBase = blockIdx.x * 128;
  const int colBase = blockIdx.y * 128;
  const int t = threadIdx.x;
  const int wave = t >> 5, lane = t & 31;
  const int laneN = lane & 15, half = lane >> 4;
  const int ldRow = t >> 1, ldSeg = (t & 1) * 16;

  auto loadTile = [&](int kt, int buf) {
    const bf16_t* ga = A  + (size_t)(rowBase + ldRow)*C_IN + kt*32 + ldSeg;
    async_cp16(&As[buf][ldRow][ldSeg],     ga);
    async_cp16(&As[buf][ldRow][ldSeg + 8], ga + 8);
    const bf16_t* gb = Bt + (size_t)(colBase + ldRow)*C_IN + kt*32 + ldSeg;
    async_cp16(&Bs[buf][ldRow][ldSeg],     gb);
    async_cp16(&Bs[buf][ldRow][ldSeg + 8], gb + 8);
  };

  v8f acc[8];
#pragma unroll
  for (int i = 0; i < 8; ++i) acc[i] = v8f_zero();

  loadTile(0, 0);
  wait_async();
  __syncthreads();

  for (int kt = 0; kt < C_IN/32; ++kt) {
    const int buf = kt & 1;
    if (kt + 1 < C_IN/32) loadTile(kt + 1, buf ^ 1);   // overlap with WMMA below

    v16bf af = frag_a(&As[buf][wave*16 + laneN][0], 0, half);
#pragma unroll
    for (int nf = 0; nf < 8; ++nf) {
      v16bf bfrag = frag_b(&Bs[buf][nf*16 + laneN][0], 0, half);
      acc[nf] = wmma_bf16(af, bfrag, acc[nf]);
    }

    wait_async();
    __syncthreads();
  }

#pragma unroll
  for (int nf = 0; nf < 8; ++nf) {
    const int gcol = colBase + nf*16 + laneN;
#pragma unroll
    for (int r = 0; r < 8; ++r) {
      const int grow = rowBase + wave*16 + half*8 + r;
      const float v = acc[nf][r] * scale;
      if (mode == 0) {
        const int b = grow >> 10, s = grow & 1023;
        const int h = gcol >> 6,  c = gcol & 63;
        ((bf16_t*)outp)[((size_t)((b*NHEAD + h) << 10) + s)*HDIM + c] = (__bf16)v;
      } else {
        ((float*)outp)[(size_t)grow*C_IN + gcol] = v + obias[gcol];
      }
    }
  }
}

// ---------------------------------------------------------------------------
// Flash attention over one (b,h,q-tile of 64): 128 threads = 4 waves,
// each wave owns 16 q-rows. Online softmax, bias + mask streamed once.
// ---------------------------------------------------------------------------
__global__ __launch_bounds__(128)
void k_attn(const bf16_t* __restrict__ Q, const bf16_t* __restrict__ K,
            const bf16_t* __restrict__ V, const float* __restrict__ bias,
            const int* __restrict__ mask, bf16_t* __restrict__ obuf) {
  __shared__ bf16_t Qs[64][72];
  __shared__ bf16_t Ks[64][72];
  __shared__ bf16_t Vt[64][72];        // transposed V: [c][kpos]
  __shared__ bf16_t Ps[4][16][72];     // wave-private P tiles
  __shared__ float  moffs[SEQ];

  const int qt = blockIdx.x & 15;      // q tile index (S/64 = 16)
  const int bh = blockIdx.x >> 4;      // b*16 + h
  const int b  = bh >> 4;
  const int h  = bh & 15;
  const int t = threadIdx.x;
  const int wave = t >> 5, lane = t & 31;
  const int laneN = lane & 15, half = lane >> 4;
  const int ldRow = t >> 1, ldSeg = (t & 1) * 32;

  // Load Q tile (already scaled by 1/sqrt(c)) asynchronously + mask offsets.
  {
    const bf16_t* gq = Q + ((size_t)(bh << 10) + qt*64 + ldRow)*HDIM + ldSeg;
#pragma unroll
    for (int j = 0; j < 4; ++j) async_cp16(&Qs[ldRow][ldSeg + 8*j], gq + 8*j);
    for (int i = t; i < SEQ; i += 128)
      moffs[i] = (mask[(b << 10) + i] == 0) ? -1e8f : 0.f;
  }

  float m[8], l[8];
#pragma unroll
  for (int r = 0; r < 8; ++r) { m[r] = -3.0e38f; l[r] = 0.f; }
  v8f o[4];
#pragma unroll
  for (int cf = 0; cf < 4; ++cf) o[cf] = v8f_zero();

  const int qrow = qt*64 + wave*16 + half*8;               // + r
  const size_t biasRow = ((size_t)bh << 20) + ((size_t)qrow << 10);

  for (int kt = 0; kt < SEQ/64; ++kt) {
    __syncthreads();
    {
      // K tile: async straight into LDS (row-major, B-fragment friendly)
      const bf16_t* gk = K + ((size_t)(bh << 10) + kt*64 + ldRow)*HDIM + ldSeg;
#pragma unroll
      for (int j = 0; j < 4; ++j) async_cp16(&Ks[ldRow][ldSeg + 8*j], gk + 8*j);
      // V tile: needs transpose -> VGPR round trip
      const v8bf* sv = (const v8bf*)(V + ((size_t)(bh << 10) + kt*64 + ldRow)*HDIM + ldSeg);
#pragma unroll
      for (int j = 0; j < 4; ++j) {
        v8bf ch = sv[j];
#pragma unroll
        for (int e = 0; e < 8; ++e) Vt[ldSeg + 8*j + e][ldRow] = ch[e];
      }
    }
    wait_async();
    __syncthreads();

    // S = Q * K^T  (contraction over c = 64, two 32-wide WMMA steps)
    v8f sf[4];
#pragma unroll
    for (int nf = 0; nf < 4; ++nf) {
      v8f a = v8f_zero();
#pragma unroll
      for (int cs = 0; cs < 2; ++cs) {
        v16bf qa = frag_a(&Qs[wave*16 + laneN][0], cs*32, half);
        v16bf kb = frag_b(&Ks[nf*16 + laneN][0], cs*32, half);
        a = wmma_bf16(qa, kb, a);
      }
      sf[nf] = a;
    }

    // + bias + mask, track new row max
    float mnew[8];
#pragma unroll
    for (int r = 0; r < 8; ++r) mnew[r] = m[r];
#pragma unroll
    for (int nf = 0; nf < 4; ++nf) {
      const int gk = kt*64 + nf*16 + laneN;
      const float moff = moffs[gk];
#pragma unroll
      for (int r = 0; r < 8; ++r) {
        float sc = sf[nf][r] + bias[biasRow + ((size_t)r << 10) + gk] + moff;
        sf[nf][r] = sc;
        mnew[r] = fmaxf(mnew[r], sc);
      }
    }
    if (kt + 1 < SEQ/64)
      __builtin_prefetch(bias + biasRow + (size_t)(kt + 1)*64, 0, 0);

    // reduce row max across the 16-lane N group (lanes 0-15 / 16-31 separate)
#pragma unroll
    for (int r = 0; r < 8; ++r) {
#pragma unroll
      for (int d = 1; d < 16; d <<= 1)
        mnew[r] = fmaxf(mnew[r], __shfl_xor(mnew[r], d, 32));
    }

    float lsum[8];
#pragma unroll
    for (int r = 0; r < 8; ++r) lsum[r] = 0.f;
#pragma unroll
    for (int nf = 0; nf < 4; ++nf)
#pragma unroll
      for (int r = 0; r < 8; ++r) {
        float p = __expf(sf[nf][r] - mnew[r]);
        sf[nf][r] = p;
        lsum[r] += p;
      }
#pragma unroll
    for (int r = 0; r < 8; ++r) {
#pragma unroll
      for (int d = 1; d < 16; d <<= 1)
        lsum[r] += __shfl_xor(lsum[r], d, 32);
      const float alpha = __expf(m[r] - mnew[r]);
      l[r] = l[r]*alpha + lsum[r];
      m[r] = mnew[r];
#pragma unroll
      for (int cf = 0; cf < 4; ++cf) o[cf][r] *= alpha;
    }

    // P -> wave-private LDS (C/D layout -> A layout round-trip)
#pragma unroll
    for (int nf = 0; nf < 4; ++nf)
#pragma unroll
      for (int r = 0; r < 8; ++r)
        Ps[wave][half*8 + r][nf*16 + laneN] = (__bf16)sf[nf][r];
    // same-wave DS is in-order; explicit split-counter wait keeps the compiler
    // from hoisting the fragment loads above the stores.
    asm volatile("s_wait_dscnt 0" ::: "memory");

    // O += P * V (contraction over kpos = 64, two 32-wide WMMA steps)
#pragma unroll
    for (int cf = 0; cf < 4; ++cf)
#pragma unroll
      for (int ks = 0; ks < 2; ++ks) {
        v16bf pa = frag_a(&Ps[wave][laneN][0], ks*32, half);
        v16bf vb = frag_b(&Vt[cf*16 + laneN][0], ks*32, half);
        o[cf] = wmma_bf16(pa, vb, o[cf]);
      }
  }

  // normalize and write merged-head bf16 (B*S, H*c)
#pragma unroll
  for (int r = 0; r < 8; ++r) {
    const float inv = 1.f / l[r];
#pragma unroll
    for (int cf = 0; cf < 4; ++cf) {
      const int col = h*HDIM + cf*16 + laneN;
      obuf[((size_t)(b << 10) + qrow + r)*C_IN + col] = (__bf16)(o[cf][r] * inv);
    }
  }
}

// ---------------------------------------------------------------------------
// launcher
// ---------------------------------------------------------------------------
extern "C" void kernel_launch(void* const* d_in, const int* in_sizes, int n_in,
                              void* d_out, int out_size, void* d_ws, size_t ws_size,
                              hipStream_t stream) {
  const float* x    = (const float*)d_in[0];
  const float* bias = (const float*)d_in[1];
  const int*   mask = (const int*)d_in[2];
  const float* Wq   = (const float*)d_in[3];
  const float* Wk   = (const float*)d_in[4];
  const float* Wv   = (const float*)d_in[5];
  const float* Wo   = (const float*)d_in[6];
  const float* bo   = (const float*)d_in[7];
  float* out = (float*)d_out;

  char* ws = (char*)d_ws;
  bf16_t* xb  = (bf16_t*)(ws + ((size_t)0  << 20));  // 8 MB
  bf16_t* Wqt = (bf16_t*)(ws + ((size_t)8  << 20));  // 2 MB each
  bf16_t* Wkt = (bf16_t*)(ws + ((size_t)10 << 20));
  bf16_t* Wvt = (bf16_t*)(ws + ((size_t)12 << 20));
  bf16_t* Wot = (bf16_t*)(ws + ((size_t)14 << 20));
  bf16_t* Qb  = (bf16_t*)(ws + ((size_t)16 << 20));  // 8 MB each
  bf16_t* Kb  = (bf16_t*)(ws + ((size_t)24 << 20));
  bf16_t* Vb  = (bf16_t*)(ws + ((size_t)32 << 20));
  bf16_t* Ob  = (bf16_t*)(ws + ((size_t)40 << 20));

  k_cvt_x<<<MTOT*C_IN/(256*4), 256, 0, stream>>>(x, xb);
  k_transpose_cvt<<<1024, 256, 0, stream>>>(Wq, Wqt);
  k_transpose_cvt<<<1024, 256, 0, stream>>>(Wk, Wkt);
  k_transpose_cvt<<<1024, 256, 0, stream>>>(Wv, Wvt);
  k_transpose_cvt<<<1024, 256, 0, stream>>>(Wo, Wot);

  dim3 gg(MTOT/128, C_IN/128, 1);
  k_gemm_bf16<<<gg, 256, 0, stream>>>(xb, Wqt, (void*)Qb, nullptr, 0.125f, 0);
  k_gemm_bf16<<<gg, 256, 0, stream>>>(xb, Wkt, (void*)Kb, nullptr, 1.0f,   0);
  k_gemm_bf16<<<gg, 256, 0, stream>>>(xb, Wvt, (void*)Vb, nullptr, 1.0f,   0);

  k_attn<<<BATCH*NHEAD*(SEQ/64), 128, 0, stream>>>(Qb, Kb, Vb, bias, mask, Ob);

  k_gemm_bf16<<<gg, 256, 0, stream>>>(Ob, Wot, (void*)out, bo, 1.0f, 1);
}
